// GPT2_28647431864395
// MI455X (gfx1250) — matrix operand order repaired and checked by reading, not compiled, not run on hardware
//
#include <hip/hip_runtime.h>
#include <hip/hip_bf16.h>
#include <math.h>

// ---------------------------------------------------------------------------
// GPT-2 small forward (B=2, T=1024, E=768, H=12, L=6, R=4, V=50257), f32 I/O.
// GEMMs: v_wmma_f32_16x16x32_f16, f32 accumulate, LDS tiles in fragment
// order (ds_load_b128 operands), double-buffered LDS software pipeline.
// Requirements relied on by the GEMM: M % 64 == 0, K % 32 == 0 (true for all
// launches here: M=2048, K in {768, 3072}).
// ---------------------------------------------------------------------------

typedef __attribute__((ext_vector_type(16))) _Float16 v16h;
typedef __attribute__((ext_vector_type(8)))  _Float16 v8h;
typedef __attribute__((ext_vector_type(4)))  _Float16 v4h;
typedef __attribute__((ext_vector_type(8)))  float    v8f;
typedef __attribute__((ext_vector_type(4)))  float    v4f;

namespace cfg {
constexpr int Bv = 2;
constexpr int Tv = 1024;
constexpr int Ev = 768;
constexpr int Hv = 12;
constexpr int Lv = 6;
constexpr int Dv = 64;          // E / H
constexpr int Vv = 50257;
constexpr int Mv = Bv * Tv;     // 2048 activation rows
}

// ---------------------------------------------------------------------------
// Embedding: h[b,t,:] = tok_emb[x[b,t]] + pos_emb[t]
// ---------------------------------------------------------------------------
__global__ void embed_kernel(const int* __restrict__ x,
                             const float* __restrict__ tok,
                             const float* __restrict__ pos,
                             float* __restrict__ h) {
    int row = blockIdx.x;              // 0 .. B*T-1
    int t   = row % cfg::Tv;
    int id  = x[row];
    const float* te = tok + (size_t)id * cfg::Ev;
    const float* pe = pos + (size_t)t  * cfg::Ev;
    float* hr = h + (size_t)row * cfg::Ev;
    for (int i = threadIdx.x; i < cfg::Ev; i += blockDim.x)
        hr[i] = te[i] + pe[i];
}

// ---------------------------------------------------------------------------
// LayerNorm: one 256-thread block per row (E = 768)
// ---------------------------------------------------------------------------
__global__ void ln_kernel(const float* __restrict__ x,
                          const float* __restrict__ gamma,
                          const float* __restrict__ beta,
                          float* __restrict__ y) {
    __shared__ float redS[8];
    __shared__ float redQ[8];
    int row = blockIdx.x;
    const float* xr = x + (size_t)row * cfg::Ev;
    float s = 0.f, q = 0.f;
    for (int i = threadIdx.x; i < cfg::Ev; i += blockDim.x) {
        float v = xr[i];
        s += v; q += v * v;
    }
    #pragma unroll
    for (int off = 16; off > 0; off >>= 1) {
        s += __shfl_xor(s, off, 32);
        q += __shfl_xor(q, off, 32);
    }
    int wid = threadIdx.x >> 5, lane = threadIdx.x & 31;
    if (lane == 0) { redS[wid] = s; redQ[wid] = q; }
    __syncthreads();
    if (threadIdx.x == 0) {
        float a = 0.f, b = 0.f;
        #pragma unroll
        for (int i = 0; i < 8; ++i) { a += redS[i]; b += redQ[i]; }
        redS[0] = a; redQ[0] = b;
    }
    __syncthreads();
    float mean = redS[0] * (1.0f / cfg::Ev);
    float var  = redQ[0] * (1.0f / cfg::Ev) - mean * mean;
    float inv  = rsqrtf(var + 1e-5f);
    float* yr = y + (size_t)row * cfg::Ev;
    for (int i = threadIdx.x; i < cfg::Ev; i += blockDim.x)
        yr[i] = (xr[i] - mean) * inv * gamma[i] + beta[i];
}

// ---------------------------------------------------------------------------
// WMMA GEMM: C[M,N] = epilogue(A[M,K] @ W[K,N] + bias [+ resid])
//
//   Block: 256 threads = 8 waves, tile 64(M) x 128(N), BK = 32.
//   Wave (wm = w&3, wn = w>>2) owns rows wm*16..+16, cols wn*64..+64
//     => 4x v_wmma_f32_16x16x32_f16 per k-step per wave.
//
//   LDS tiles stored in WMMA *fragment order* (ISA 7.12.2):
//     A 16x32 f16 frag : lane = m + (k&8 ? 16:0), elem = (k&7)|((k&16)>>1)
//     B 32x16 f16 frag : lane = n + (k&8 ? 16:0), elem = (k&7)|((k&16)>>1)
//   -> per-lane operand is contiguous 32B: 2x ds_load_b128.
//   Lane stride 24 f16 (48B): 16B-aligned, breaks 4-way bank conflicts.
//
//   Double-buffered software pipeline: global loads for tile k+1 issue
//   before the WMMAs of tile k; one s_barrier per iteration.
//
//   Preconditions: M % 64 == 0, K % 32 == 0. N may be ragged (head: 50257).
//   act: 0 = none, 1 = exact GELU.
// ---------------------------------------------------------------------------
__global__ void gemm_wmma_kernel(const float* __restrict__ A,
                                 const float* __restrict__ W,
                                 const float* __restrict__ bias,
                                 const float* __restrict__ resid,
                                 float* __restrict__ C,
                                 int M, int N, int K, int act) {
    constexpr int BM = 64, BN = 128, BK = 32;
    constexpr int LSTR = 24;                    // f16 per lane slot (16 + 8 pad)
    __shared__ _Float16 Af[2][4][32][LSTR];     // [buf][m-strip][lane][elem]
    __shared__ _Float16 Bf[2][8][32][LSTR];     // [buf][n-tile ][lane][elem]

    const int tid  = threadIdx.x;
    const int lane = tid & 31;
    const int wave = tid >> 5;
    const int wm   = wave & 3;                  // 16-row strip
    const int wn   = wave >> 2;                 // 64-col strip (4 n-tiles)
    const int mbase = blockIdx.y * BM;
    const int nbase = blockIdx.x * BN;

    const int  lm = lane & 15;
    const bool hi = lane >= 16;
    const int  nk = K / BK;

    v8f acc[4];
    #pragma unroll
    for (int t = 0; t < 4; ++t)
        acc[t] = (v8f){0.f,0.f,0.f,0.f,0.f,0.f,0.f,0.f};

    v4f aR[2];   // staged A quads (rows always in range: M % 64 == 0)
    v4f bR[4];   // staged B quads (ragged N guarded)

    // ---- stage tile k0 from global into registers -------------------------
    auto load_tile = [&](int k0) {
        #pragma unroll
        for (int i = 0; i < 2; ++i) {
            const int idx = tid + 256 * i;      // 0..511
            const int row = idx >> 3;           // 0..63
            const int qk  = (idx & 7) * 4;      // 0,4,...,28
            const float* ap = A + (size_t)(mbase + row) * K + (k0 + qk);
            aR[i] = *(const v4f*)ap;
            if (k0 + BK < K)
                __builtin_prefetch(ap + BK, 0, 1);
        }
        #pragma unroll
        for (int i = 0; i < 4; ++i) {
            const int idx  = tid + 256 * i;     // 0..1023
            const int kq   = idx >> 7;          // 0..7  (k quad)
            const int nloc = idx & 127;         // 0..127
            const int gn   = nbase + nloc;
            v4f f = (v4f){0.f, 0.f, 0.f, 0.f};
            if (gn < N) {
                const float* wp = W + (size_t)(k0 + kq * 4) * N + gn;
                f.x = wp[0];
                f.y = wp[(size_t)N];
                f.z = wp[2 * (size_t)N];
                f.w = wp[3 * (size_t)N];
                if (k0 + BK < K)
                    __builtin_prefetch(wp + (size_t)BK * N, 0, 1);
            }
            bR[i] = f;
        }
    };

    // ---- convert f32->f16, scatter to fragment-order LDS ------------------
    auto store_tile = [&](int buf) {
        #pragma unroll
        for (int i = 0; i < 2; ++i) {
            const int idx = tid + 256 * i;
            const int row = idx >> 3;
            const int qk  = (idx & 7) * 4;
            v4h hq = { (_Float16)aR[i].x, (_Float16)aR[i].y,
                       (_Float16)aR[i].z, (_Float16)aR[i].w };
            const int strip = row >> 4;
            const int lt    = (row & 15) + ((qk & 8) ? 16 : 0);
            const int e0    = (qk & 7) | ((qk & 16) >> 1);
            *(v4h*)&Af[buf][strip][lt][e0] = hq;
        }
        #pragma unroll
        for (int i = 0; i < 4; ++i) {
            const int idx  = tid + 256 * i;
            const int kq   = idx >> 7;
            const int nloc = idx & 127;
            v4h hq = { (_Float16)bR[i].x, (_Float16)bR[i].y,
                       (_Float16)bR[i].z, (_Float16)bR[i].w };
            const int ntile = nloc >> 4;
            const int lt    = (nloc & 15) + ((kq & 2) ? 16 : 0);  // k&8
            const int e0    = ((kq * 4) & 7) | (((kq * 4) & 16) >> 1);
            *(v4h*)&Bf[buf][ntile][lt][e0] = hq;
        }
    };

    // ---- pipeline ---------------------------------------------------------
    load_tile(0);
    store_tile(0);
    __syncthreads();

    for (int kt = 0; kt < nk; ++kt) {
        const int cur = kt & 1;
        const bool more = (kt + 1) < nk;
        if (more)
            load_tile((kt + 1) * BK);           // global loads overlap WMMAs

        const v8h a0 = *(const v8h*)&Af[cur][wm][lane][0];
        const v8h a1 = *(const v8h*)&Af[cur][wm][lane][8];
        const v16h af = __builtin_shufflevector(a0, a1,
            0,1,2,3,4,5,6,7,8,9,10,11,12,13,14,15);

        #pragma unroll
        for (int t = 0; t < 4; ++t) {
            const int nt = wn * 4 + t;
            const v8h b0 = *(const v8h*)&Bf[cur][nt][lane][0];
            const v8h b1 = *(const v8h*)&Bf[cur][nt][lane][8];
            const v16h bf = __builtin_shufflevector(b0, b1,
                0,1,2,3,4,5,6,7,8,9,10,11,12,13,14,15);
            acc[t] = __builtin_amdgcn_wmma_f32_16x16x32_f16(
                false, af, false, bf, (short)0, acc[t], false, false);
        }

        if (more)
            store_tile(cur ^ 1);                // fill the other buffer
        __syncthreads();                        // one barrier per k-step
    }

    // ---- epilogue: bias, activation, residual, guarded store --------------
    #pragma unroll
    for (int t = 0; t < 4; ++t) {
        const int gn = nbase + (wn * 4 + t) * 16 + lm;
        if (gn >= N) continue;
        const float bv = bias ? bias[gn] : 0.f;
        #pragma unroll
        for (int r = 0; r < 8; ++r) {
            const int gm = mbase + wm * 16 + r + (hi ? 8 : 0);
            float v = acc[t][r] + bv;
            if (act == 1)  // exact GELU: 0.5*x*(1+erf(x/sqrt(2)))
                v = 0.5f * v * (1.f + erff(v * 0.70710678118654752f));
            if (resid) v += resid[(size_t)gm * N + gn];
            C[(size_t)gm * N + gn] = v;
        }
    }
}

// ---------------------------------------------------------------------------
// Causal attention, flash-style online softmax.
// One wave per (b, h, q) row. D=64 -> each lane owns dims {lane, lane+32}.
// qkv layout: [B*T, 3E] with q | k | v slabs of width E.
// ---------------------------------------------------------------------------
__global__ void attn_kernel(const float* __restrict__ qkv,
                            float* __restrict__ out) {
    const int wave = (blockIdx.x * blockDim.x + threadIdx.x) >> 5;
    const int lane = threadIdx.x & 31;
    const int q  = wave % cfg::Tv;
    const int bh = wave / cfg::Tv;
    const int h  = bh % cfg::Hv;
    const int b  = bh / cfg::Hv;

    const size_t rowstride = 3 * (size_t)cfg::Ev;
    const float* qptr = qkv + ((size_t)(b * cfg::Tv + q)) * rowstride + h * cfg::Dv;
    const float q0 = qptr[lane];
    const float q1 = qptr[lane + 32];

    float m = -3.0e38f, l = 0.f, a0 = 0.f, a1 = 0.f;
    const float scale = 0.125f;  // 1/sqrt(64)

    for (int k = 0; k <= q; ++k) {
        const float* kp = qkv + ((size_t)(b * cfg::Tv + k)) * rowstride
                          + cfg::Ev + h * cfg::Dv;
        float part = q0 * kp[lane] + q1 * kp[lane + 32];
        #pragma unroll
        for (int off = 16; off > 0; off >>= 1)
            part += __shfl_xor(part, off, 32);
        const float s    = part * scale;
        const float mnew = fmaxf(m, s);
        const float corr = __expf(m - mnew);
        const float p    = __expf(s - mnew);
        const float* vp  = qkv + ((size_t)(b * cfg::Tv + k)) * rowstride
                           + 2 * cfg::Ev + h * cfg::Dv;
        a0 = a0 * corr + p * vp[lane];
        a1 = a1 * corr + p * vp[lane + 32];
        l  = l  * corr + p;
        m  = mnew;
    }
    float* op = out + ((size_t)(b * cfg::Tv + q)) * cfg::Ev + h * cfg::Dv;
    const float inv = 1.f / l;
    op[lane]      = a0 * inv;
    op[lane + 32] = a1 * inv;
}

// ---------------------------------------------------------------------------
// Launch
// ---------------------------------------------------------------------------
static inline dim3 gemm_grid(int M, int N) {
    return dim3((unsigned)((N + 127) / 128), (unsigned)((M + 63) / 64), 1);
}

extern "C" void kernel_launch(void* const* d_in, const int* in_sizes, int n_in,
                              void* d_out, int out_size, void* d_ws, size_t ws_size,
                              hipStream_t stream) {
    using namespace cfg;
    (void)in_sizes; (void)n_in; (void)out_size; (void)ws_size;

    const int*   x       = (const int*)  d_in[0];
    const float* tok_emb = (const float*)d_in[1];
    const float* pos_emb = (const float*)d_in[2];
    const float* ln1_w   = (const float*)d_in[3];
    const float* ln1_b   = (const float*)d_in[4];
    const float* qkv_w   = (const float*)d_in[5];
    const float* qkv_b   = (const float*)d_in[6];
    const float* out_w   = (const float*)d_in[7];
    const float* out_b   = (const float*)d_in[8];
    const float* ln2_w   = (const float*)d_in[9];
    const float* ln2_b   = (const float*)d_in[10];
    const float* mlp_w1  = (const float*)d_in[11];
    const float* mlp_b1  = (const float*)d_in[12];
    const float* mlp_w2  = (const float*)d_in[13];
    const float* mlp_b2  = (const float*)d_in[14];
    const float* lnf_w   = (const float*)d_in[15];
    const float* lnf_b   = (const float*)d_in[16];
    const float* head_w  = (const float*)d_in[17];
    float* logits = (float*)d_out;

    // workspace carve (floats)
    float* ws   = (float*)d_ws;
    float* h    = ws;                         // [M, E]
    float* hn   = h   + (size_t)Mv * Ev;      // [M, E]
    float* qkv  = hn  + (size_t)Mv * Ev;      // [M, 3E]
    float* attn = qkv + (size_t)Mv * 3 * Ev;  // [M, E]
    float* mlp  = attn + (size_t)Mv * Ev;     // [M, 4E]

    const dim3 blk(256);

    // token + position embedding
    embed_kernel<<<dim3(Mv), blk, 0, stream>>>(x, tok_emb, pos_emb, h);

    for (int l = 0; l < Lv; ++l) {
        const float* l1w = ln1_w + (size_t)l * Ev;
        const float* l1b = ln1_b + (size_t)l * Ev;
        const float* qw  = qkv_w + (size_t)l * Ev * 3 * Ev;
        const float* qb  = qkv_b + (size_t)l * 3 * Ev;
        const float* ow  = out_w + (size_t)l * Ev * Ev;
        const float* ob  = out_b + (size_t)l * Ev;
        const float* l2w = ln2_w + (size_t)l * Ev;
        const float* l2b = ln2_b + (size_t)l * Ev;
        const float* w1  = mlp_w1 + (size_t)l * Ev * 4 * Ev;
        const float* b1  = mlp_b1 + (size_t)l * 4 * Ev;
        const float* w2  = mlp_w2 + (size_t)l * 4 * Ev * Ev;
        const float* b2  = mlp_b2 + (size_t)l * Ev;

        // hn = LN1(h)
        ln_kernel<<<dim3(Mv), blk, 0, stream>>>(h, l1w, l1b, hn);
        // qkv = hn @ qw + qb
        gemm_wmma_kernel<<<gemm_grid(Mv, 3 * Ev), blk, 0, stream>>>(
            hn, qw, qb, nullptr, qkv, Mv, 3 * Ev, Ev, 0);
        // attn = softmax(q k^T / sqrt(D), causal) @ v
        attn_kernel<<<dim3(Bv * Hv * Tv / 8), blk, 0, stream>>>(qkv, attn);
        // h = h + attn @ ow + ob
        gemm_wmma_kernel<<<gemm_grid(Mv, Ev), blk, 0, stream>>>(
            attn, ow, ob, h, h, Mv, Ev, Ev, 0);
        // hn = LN2(h)
        ln_kernel<<<dim3(Mv), blk, 0, stream>>>(h, l2w, l2b, hn);
        // mlp = gelu(hn @ w1 + b1)
        gemm_wmma_kernel<<<gemm_grid(Mv, 4 * Ev), blk, 0, stream>>>(
            hn, w1, b1, nullptr, mlp, Mv, 4 * Ev, Ev, 1);
        // h = h + mlp @ w2 + b2
        gemm_wmma_kernel<<<gemm_grid(Mv, Ev), blk, 0, stream>>>(
            mlp, w2, b2, h, h, Mv, Ev, 4 * Ev, 0);
    }

    // final LN + head
    ln_kernel<<<dim3(Mv), blk, 0, stream>>>(h, lnf_w, lnf_b, hn);
    gemm_wmma_kernel<<<gemm_grid(Mv, Vv), blk, 0, stream>>>(
        hn, head_w, nullptr, nullptr, logits, Mv, Vv, Ev, 0);
}
